// CausalAttention_15255723835643
// MI455X (gfx1250) — matrix operand order repaired and checked by reading, not compiled, not run on hardware
//
#include <hip/hip_runtime.h>

typedef __attribute__((ext_vector_type(16))) _Float16 v16h;
typedef __attribute__((ext_vector_type(8)))  _Float16 v8h;
typedef __attribute__((ext_vector_type(4)))  _Float16 v4h;
typedef __attribute__((ext_vector_type(8)))  float    v8f;

#define DEV __device__ __forceinline__

#ifndef USE_ASYNC_LDS
#define USE_ASYNC_LDS 1
#endif

// ---------------------------------------------------------------------------
// CDNA5 async global->LDS copy (GLOBAL_LOAD_ASYNC_TO_LDS_B128, ASYNCcnt).
// GVS addressing: uniform SGPR64 base + per-lane 32-bit byte offset.
// LDS destination = low 32 bits of the generic __shared__ pointer.
// ---------------------------------------------------------------------------
DEV void copy_b128(_Float16* lds_p, const void* base, unsigned goff) {
#if USE_ASYNC_LDS
  asm volatile("global_load_async_to_lds_b128 %0, %1, %2"
               :
               : "v"((unsigned)(size_t)lds_p), "v"(goff), "s"(base)
               : "memory");
#else
  *(uint4*)lds_p = *(const uint4*)((const char*)base + goff);
#endif
}
DEV void copy_wait_group() {  // previous tile's 4 copies done (4 newer pending)
#if USE_ASYNC_LDS
  asm volatile("s_wait_asynccnt 0x4" ::: "memory");
#endif
}
DEV void copy_wait_all() {
#if USE_ASYNC_LDS
  asm volatile("s_wait_asynccnt 0x0" ::: "memory");
#endif
}

// ---------------------------------------------------------------------------
// WMMA fragment helpers (CDNA5 wave32, V_WMMA_F32_16X16X32_F16 layouts)
// A 16x32 f16: lane%16 = M row; 16 contiguous K halves, +16 for lanes>=16
// (same K order as B).   (this code's own order, not the ISA table order)
// B 32x16 f16: lane%16 = N col; 16 contiguous K halves, +16 for lanes>=16.
// C/D 16x16 f32: lane%16 = N; vgpr r = M row (+8 for lanes>=16).
// ---------------------------------------------------------------------------
DEV v16h frag_a(const _Float16* base, int row, int lane) {
  const _Float16* p = base + row * 32;
  const int k0 = (lane & 16) ? 16 : 0;
  return *(const v16h*)(p + k0);
}
DEV v16h frag_b(const _Float16* base, int row, int lane) {
  return *(const v16h*)(base + row * 32 + ((lane & 16) ? 16 : 0));
}
DEV v8f wmma_f16(v16h a, v16h b, v8f c) {
  return __builtin_amdgcn_wmma_f32_16x16x32_f16(false, a, false, b, (short)0, c,
                                                false, false);
}

// ---------------------------------------------------------------------------
// fp32 -> fp16 elementwise convert (4 elems/thread)
// ---------------------------------------------------------------------------
__global__ __launch_bounds__(256) void cvt_f32_f16(const float* __restrict__ in,
                                                   _Float16* __restrict__ out,
                                                   long n) {
  long i = ((long)blockIdx.x * 256 + threadIdx.x) * 4;
  if (i + 3 < n) {
    float4 v = *(const float4*)(in + i);
    v4h h;
    h[0] = (_Float16)v.x; h[1] = (_Float16)v.y;
    h[2] = (_Float16)v.z; h[3] = (_Float16)v.w;
    *(v4h*)(out + i) = h;
  }
}

// ---------------------------------------------------------------------------
// fp32 [R][C] -> fp16 [C][R] transpose-convert (32x32 LDS tiles)
// ---------------------------------------------------------------------------
__global__ __launch_bounds__(256) void transpose_cvt(const float* __restrict__ in,
                                                     _Float16* __restrict__ out,
                                                     int R, int C) {
  __shared__ float tile[32][33];
  const int c0 = blockIdx.x * 32, r0 = blockIdx.y * 32;
  for (int i = threadIdx.y; i < 32; i += 8)
    tile[i][threadIdx.x] = in[(size_t)(r0 + i) * C + c0 + threadIdx.x];
  __syncthreads();
  for (int i = threadIdx.y; i < 32; i += 8)
    out[(size_t)(c0 + i) * R + r0 + threadIdx.x] = (_Float16)tile[threadIdx.x][i];
}

// ---------------------------------------------------------------------------
// WMMA GEMM: D[M,N] = A[M,K] * Bt[N,K]^T     (A,Bt f16 row-major, K-contig)
// Block 256 thr = 8 waves (4 over M x 2 over N); tile 128x128x32.
// Peeled async double-buffered pipeline: steady-state loop is branch-free:
//   issue(k+1 -> buf^1); s_wait_asynccnt 4; barrier; frag loads; 8x wmma;
//   barrier.  All fragment ds_loads hoisted ahead of the WMMA burst.
// MODE 0: f16 row-major out     (Q, K, O)
// MODE 1: f16 transposed out Vt[b][n][t] (n=N index, t=m%T), batch b=m/T
// MODE 2: f32 row-major out + bias     (final projection)
// ---------------------------------------------------------------------------
template <int MODE>
__global__ __launch_bounds__(256) void gemm_kernel(
    const _Float16* __restrict__ A, const _Float16* __restrict__ Bt,
    void* __restrict__ outp, const float* __restrict__ bias,
    int M, int N, int K, long sA, long sB, long sO, int T) {
  __shared__ _Float16 As[2][128 * 32];
  __shared__ _Float16 Bs[2][128 * 32];
  const int tid = threadIdx.x, lane = tid & 31, w = tid >> 5;
  const int wm = w & 3, wn = w >> 2;
  A  += (size_t)blockIdx.z * sA;
  Bt += (size_t)blockIdx.z * sB;

  const v8f vzero = {0.f, 0.f, 0.f, 0.f, 0.f, 0.f, 0.f, 0.f};
  v8f acc[2][4];
#pragma unroll
  for (int i = 0; i < 2; i++)
#pragma unroll
    for (int j = 0; j < 4; j++) acc[i][j] = vzero;

  const int lr = tid >> 1, lch = (tid & 1) * 16;
  const unsigned aOff = (unsigned)((((size_t)blockIdx.y * 128 + lr) * K + lch) * 2);
  const unsigned bOff = (unsigned)((((size_t)blockIdx.x * 128 + lr) * K + lch) * 2);
  _Float16* asp[2] = {&As[0][lr * 32 + lch], &As[1][lr * 32 + lch]};
  _Float16* bsp[2] = {&Bs[0][lr * 32 + lch], &Bs[1][lr * 32 + lch]};

  auto issue = [&](unsigned kbytes, int pb) {
    copy_b128(asp[pb],     A,  aOff + kbytes);
    copy_b128(asp[pb] + 8, A,  aOff + kbytes + 16);
    copy_b128(bsp[pb],     Bt, bOff + kbytes);
    copy_b128(bsp[pb] + 8, Bt, bOff + kbytes + 16);
  };
  auto compute = [&](int pb) {
    v16h a0 = frag_a(&As[pb][0], wm * 32 + (lane & 15), lane);
    v16h a1 = frag_a(&As[pb][0], wm * 32 + 16 + (lane & 15), lane);
    v16h bfv[4];
#pragma unroll
    for (int j = 0; j < 4; j++)
      bfv[j] = frag_b(&Bs[pb][0], wn * 64 + j * 16 + (lane & 15), lane);
#pragma unroll
    for (int j = 0; j < 4; j++) {
      acc[0][j] = wmma_f16(a0, bfv[j], acc[0][j]);
      acc[1][j] = wmma_f16(a1, bfv[j], acc[1][j]);
    }
  };

  issue(0u, 0);
  int buf = 0;
  for (int k0 = 0; k0 + 32 < K; k0 += 32) {  // branch-free steady state
    issue((unsigned)(k0 + 32) * 2, buf ^ 1);
    copy_wait_group();
    __syncthreads();
    compute(buf);
    __syncthreads();
    buf ^= 1;
  }
  copy_wait_all();
  __syncthreads();
  compute(buf);  // drained tail; no trailing barrier needed before epilogue

  const int mBase = blockIdx.y * 128 + wm * 32;
  const int nBase = blockIdx.x * 128 + wn * 64;
#pragma unroll
  for (int i = 0; i < 2; i++) {
#pragma unroll
    for (int j = 0; j < 4; j++) {
      const int n  = nBase + j * 16 + (lane & 15);
      const int m0 = mBase + i * 16 + ((lane & 16) ? 8 : 0);
      if constexpr (MODE == 0) {
        _Float16* o = (_Float16*)outp + (size_t)blockIdx.z * sO;
#pragma unroll
        for (int r = 0; r < 8; r++)
          o[(size_t)(m0 + r) * N + n] = (_Float16)acc[i][j][r];
      } else if constexpr (MODE == 2) {
        float* o = (float*)outp;
        const float bv = bias[n];
#pragma unroll
        for (int r = 0; r < 8; r++)
          o[(size_t)(m0 + r) * N + n] = acc[i][j][r] + bv;
      } else {  // MODE 1: Vt[b][n][t], 8 contiguous t per lane -> one b128
        const int bidx = m0 / T, tloc = m0 - bidx * T;
        union { _Float16 h[8]; uint4 u4; } pk;
#pragma unroll
        for (int r = 0; r < 8; r++) pk.h[r] = (_Float16)acc[i][j][r];
        _Float16* o = (_Float16*)outp;
        *(uint4*)&o[((size_t)bidx * N + n) * T + tloc] = pk.u4;
      }
    }
  }
}

// ---------------------------------------------------------------------------
// Causal attention probabilities: P[b][t][s] = softmax(Q.K^T with tril mask)
// Block = 128 query rows (8 waves x 16 rows), grid (T/128, B).
// Phase 0: stream s-tiles 0..diag, WMMA scores, online row max/sum.
// Phase 1: recompute scores, write normalized f16 P (zeros past diagonal).
// Same peeled async double-buffered pipeline over the feature dimension.
// ---------------------------------------------------------------------------
__global__ __launch_bounds__(256) void attn_kernel(
    const _Float16* __restrict__ Q, const _Float16* __restrict__ Kx,
    _Float16* __restrict__ P, int T, int F) {
  __shared__ _Float16 Qs[2][128 * 32];
  __shared__ _Float16 Ks[2][128 * 32];
  const int tid = threadIdx.x, lane = tid & 31, w = tid >> 5;
  const int tb = blockIdx.x, b = blockIdx.y;
  const int t0 = tb * 128 + w * 16;
  const int lr = tid >> 1, lch = (tid & 1) * 16;
  const v8f vzero = {0.f, 0.f, 0.f, 0.f, 0.f, 0.f, 0.f, 0.f};

  const _Float16* Qb = Q + (size_t)b * T * F;   // uniform batch bases
  const _Float16* Kb = Kx + (size_t)b * T * F;
  const unsigned qOff = (unsigned)((((size_t)tb * 128 + lr) * F + lch) * 2);
  _Float16* qsp[2] = {&Qs[0][lr * 32 + lch], &Qs[1][lr * 32 + lch]};
  _Float16* ksp[2] = {&Ks[0][lr * 32 + lch], &Ks[1][lr * 32 + lch]};

  float m_run[8], s_run[8], inv[8];
#pragma unroll
  for (int r = 0; r < 8; r++) { m_run[r] = -3.0e38f; s_run[r] = 0.f; inv[r] = 0.f; }

  const int nSB = T / 128;
  for (int phase = 0; phase < 2; phase++) {
    const int sbEnd = phase ? (nSB - 1) : tb;
    for (int sb = 0; sb <= sbEnd; sb++) {
      const bool active = (sb <= tb);
      v8f sacc[8];
#pragma unroll
      for (int j = 0; j < 8; j++) sacc[j] = vzero;

      if (active) {
        const unsigned kOff = (unsigned)((((size_t)sb * 128 + lr) * F + lch) * 2);
        auto issue = [&](unsigned fbytes, int pb) {
          copy_b128(qsp[pb],     Qb, qOff + fbytes);
          copy_b128(qsp[pb] + 8, Qb, qOff + fbytes + 16);
          copy_b128(ksp[pb],     Kb, kOff + fbytes);
          copy_b128(ksp[pb] + 8, Kb, kOff + fbytes + 16);
        };
        auto compute = [&](int pb) {
          v16h a = frag_a(&Qs[pb][0], w * 16 + (lane & 15), lane);
          v16h bfv[8];
#pragma unroll
          for (int j = 0; j < 8; j++)
            bfv[j] = frag_b(&Ks[pb][0], j * 16 + (lane & 15), lane);
#pragma unroll
          for (int j = 0; j < 8; j++) sacc[j] = wmma_f16(a, bfv[j], sacc[j]);
        };
        issue(0u, 0);
        int buf = 0;
        for (int f0 = 0; f0 + 32 < F; f0 += 32) {  // branch-free steady state
          issue((unsigned)(f0 + 32) * 2, buf ^ 1);
          copy_wait_group();
          __syncthreads();
          compute(buf);
          __syncthreads();
          buf ^= 1;
        }
        copy_wait_all();
        __syncthreads();
        compute(buf);
        __syncthreads();  // LDS reused by the next s-block's prologue

        if (sb == tb) {  // diagonal tile: causal mask
#pragma unroll
          for (int j = 0; j < 8; j++) {
            const int s = sb * 128 + j * 16 + (lane & 15);
#pragma unroll
            for (int r = 0; r < 8; r++) {
              const int t = t0 + r + ((lane & 16) ? 8 : 0);
              if (s > t) sacc[j][r] = -3.0e38f;
            }
          }
        }
      }

      if (phase == 0) {  // online max/sum per row (row lives in a 16-lane half)
#pragma unroll
        for (int r = 0; r < 8; r++) {
          float mx = -3.0e38f;
#pragma unroll
          for (int j = 0; j < 8; j++) mx = fmaxf(mx, sacc[j][r]);
#pragma unroll
          for (int o = 1; o < 16; o <<= 1) mx = fmaxf(mx, __shfl_xor(mx, o, 32));
          const float nm = fmaxf(m_run[r], mx);
          float ss = 0.f;
#pragma unroll
          for (int j = 0; j < 8; j++) ss += __expf(sacc[j][r] - nm);
#pragma unroll
          for (int o = 1; o < 16; o <<= 1) ss += __shfl_xor(ss, o, 32);
          s_run[r] = s_run[r] * __expf(m_run[r] - nm) + ss;
          m_run[r] = nm;
        }
      } else {  // write normalized probabilities (0 past the diagonal)
#pragma unroll
        for (int j = 0; j < 8; j++) {
          const int s = sb * 128 + j * 16 + (lane & 15);
#pragma unroll
          for (int r = 0; r < 8; r++) {
            const int t = t0 + r + ((lane & 16) ? 8 : 0);
            const float p =
                active ? (__expf(sacc[j][r] - m_run[r]) * inv[r]) : 0.f;
            P[(size_t)b * T * T + (size_t)t * T + s] = (_Float16)p;
          }
        }
      }
    }
    if (phase == 0) {
#pragma unroll
      for (int r = 0; r < 8; r++) inv[r] = 1.0f / s_run[r];
    }
  }
}

// ---------------------------------------------------------------------------
// Host-side orchestration
// ---------------------------------------------------------------------------
extern "C" void kernel_launch(void* const* d_in, const int* in_sizes, int n_in,
                              void* d_out, int out_size, void* d_ws,
                              size_t ws_size, hipStream_t stream) {
  constexpr int B = 4, T = 2048, C = 1024, F = 2048, MTOT = B * T;
  const float* x  = (const float*)d_in[0];
  const float* Wq = (const float*)d_in[1];
  const float* Wk = (const float*)d_in[2];
  const float* Wv = (const float*)d_in[3];
  const float* Wf = (const float*)d_in[4];
  const float* bf = (const float*)d_in[5];
  float* out = (float*)d_out;

  char* ws = (char*)d_ws;
  const size_t MB = 1ull << 20;
  _Float16* xh  = (_Float16*)(ws);              // 16 MB  [MTOT][C]
  _Float16* WqT = (_Float16*)(ws + 16 * MB);    //  4 MB  [F][C]
  _Float16* WkT = (_Float16*)(ws + 20 * MB);    //  4 MB
  _Float16* WvT = (_Float16*)(ws + 24 * MB);    //  4 MB
  _Float16* WfT = (_Float16*)(ws + 28 * MB);    //  8 MB  [F][F]
  _Float16* Qh  = (_Float16*)(ws + 36 * MB);    // 32 MB  [MTOT][F]
  _Float16* Kh  = (_Float16*)(ws + 68 * MB);    // 32 MB  [MTOT][F]
  _Float16* Vt  = (_Float16*)(ws + 100 * MB);   // 32 MB  [B][F][T]
  _Float16* Pa  = (_Float16*)(ws + 132 * MB);   // 32 MB  [B][T][T]
  _Float16* Oh  = (_Float16*)(ws + 164 * MB);   // 32 MB  [MTOT][F]

  // 1) convert x -> f16
  cvt_f32_f16<<<(MTOT * (long)C) / 1024, 256, 0, stream>>>(x, xh, (long)MTOT * C);

  // 2) transpose-convert weights to [N][K] f16
  transpose_cvt<<<dim3(F / 32, C / 32), dim3(32, 8), 0, stream>>>(Wq, WqT, C, F);
  transpose_cvt<<<dim3(F / 32, C / 32), dim3(32, 8), 0, stream>>>(Wk, WkT, C, F);
  transpose_cvt<<<dim3(F / 32, C / 32), dim3(32, 8), 0, stream>>>(Wv, WvT, C, F);
  transpose_cvt<<<dim3(F / 32, F / 32), dim3(32, 8), 0, stream>>>(Wf, WfT, F, F);

  // 3) QKV projections (V written transposed for the later P*V GEMM)
  gemm_kernel<0><<<dim3(F / 128, MTOT / 128, 1), 256, 0, stream>>>(
      xh, WqT, Qh, nullptr, MTOT, F, C, 0, 0, 0, T);
  gemm_kernel<0><<<dim3(F / 128, MTOT / 128, 1), 256, 0, stream>>>(
      xh, WkT, Kh, nullptr, MTOT, F, C, 0, 0, 0, T);
  gemm_kernel<1><<<dim3(F / 128, MTOT / 128, 1), 256, 0, stream>>>(
      xh, WvT, Vt, nullptr, MTOT, F, C, 0, 0, 0, T);

  // 4) causal softmax probabilities
  attn_kernel<<<dim3(T / 128, B), 256, 0, stream>>>(Qh, Kh, Pa, T, F);

  // 5) O = P * V  (batched; Vt rows are f with contiguous s)
  gemm_kernel<0><<<dim3(F / 128, T / 128, B), 256, 0, stream>>>(
      Pa, Vt, Oh, nullptr, T, F, T, (long)T * T, (long)F * T, (long)T * F, T);

  // 6) out = O * Wf + bf  (f32 output)
  gemm_kernel<2><<<dim3(F / 128, MTOT / 128, 1), 256, 0, stream>>>(
      Oh, WfT, (void*)out, bf, MTOT, F, F, 0, 0, 0, T);
}
